// Virtual_pts_BEV_Warp_79688823210364
// MI455X (gfx1250) — matrix operand order repaired and checked by reading, not compiled, not run on hardware
//
#include <hip/hip_runtime.h>
#include <math.h>

typedef _Float16 h16;
typedef __attribute__((ext_vector_type(16))) _Float16 v16h;
typedef __attribute__((ext_vector_type(8)))  _Float16 v8h;
typedef __attribute__((ext_vector_type(4)))  _Float16 v4h;
typedef __attribute__((ext_vector_type(8)))  float    v8f;

#define HH   128
#define WWD  128
#define NQ   (HH*WWD)
#define DD   256
#define CINC 384
#define NHD  8
#define NPT  4
#define HDIM 32

// ---------------------------------------------------------------------------
// f32 -> f16 convert, optional transpose. in: R x C row-major.
// trans=0: out R x C.  trans=1: out C x R (out[c*R+r] = in[r*C+c]).
// ---------------------------------------------------------------------------
__global__ void k_cvt(const float* __restrict__ in, h16* __restrict__ out,
                      int R, int C, int trans) {
  int i = blockIdx.x * 256 + threadIdx.x;
  if (i >= R * C) return;
  int r = i / C, c = i - r * C;
  float v = in[i];
  if (trans) out[(size_t)c * R + r] = (h16)v;
  else       out[i] = (h16)v;
}

// ---------------------------------------------------------------------------
// Learned positional encoding: pos[n*256+d] = d<128 ? col_embed[x][d]
//                                                   : row_embed[y][d-128]
// ---------------------------------------------------------------------------
__global__ void k_pos(const float* __restrict__ row_embed,
                      const float* __restrict__ col_embed,
                      float* __restrict__ pos) {
  int i = blockIdx.x * 256 + threadIdx.x;          // over NQ*DD
  if (i >= NQ * DD) return;
  int n = i >> 8, d = i & 255;
  int y = n >> 7, x = n & 127;
  pos[i] = (d < 128) ? col_embed[x * 128 + d] : row_embed[y * 128 + (d - 128)];
}

// qp16 = (half)(q32 + pos32)
__global__ void k_addpos(const float* __restrict__ q, const float* __restrict__ pos,
                         h16* __restrict__ qp) {
  int i = blockIdx.x * 256 + threadIdx.x;
  if (i >= NQ * DD) return;
  qp[i] = (h16)(q[i] + pos[i]);
}

// ---------------------------------------------------------------------------
// WMMA GEMM:  C[M,N] = A[M,K](f16,row-major) * Bt[N,K](f16,row-major)^T
//             + bias[N] (+ residual[M,N])
// One wave (32 threads) computes one 16x16 tile with v_wmma_f32_16x16x32_f16.
// out32 (optional): f32 store, trans_out=1 -> out32[col*M+row] (BEV layout).
// out16 (optional): f16 copy for the next GEMM's A operand.
// ---------------------------------------------------------------------------
__global__ __launch_bounds__(32)
void k_gemm(const h16* __restrict__ A, const h16* __restrict__ Bt,
            const float* __restrict__ bias, const float* __restrict__ res,
            float* __restrict__ out32, h16* __restrict__ out16,
            int M, int N, int K, int trans_out) {
  const int lane = threadIdx.x;
  const int l15  = lane & 15;
  const int hi   = lane >> 4;
  const int m0   = blockIdx.x * 16;
  const int n0   = blockIdx.y * 16;

  v8f acc = {};
  const h16* aBase = A  + (size_t)(m0 + l15) * K;   // A row (M index)
  const h16* bBase = Bt + (size_t)(n0 + l15) * K;   // B column (N index)

  for (int kb = 0; kb < K; kb += 32) {
    // A 16x32 f16 fragment: lanes 0-15 hold K {kb..kb+7, kb+16..kb+23},
    // lanes 16-31 hold K {kb+8..kb+15, kb+24..kb+31}   (ISA 7.12.2)
    const h16* ap = aBase + kb + hi * 8;
    v8h a0 = *(const v8h*)(ap);
    v8h a1 = *(const v8h*)(ap + 16);
    // B 32x16 fragment: same K order as A (lanes 0-15 {kb..kb+7, kb+16..kb+23}, lanes 16-31 {kb+8..kb+15, kb+24..kb+31})
    const h16* bp = bBase + kb + hi * 8;
    v8h b0 = *(const v8h*)(bp);
    v8h b1 = *(const v8h*)(bp + 16);
    v16h av, bv;
#pragma unroll
    for (int e = 0; e < 8; ++e) {
      av[e] = a0[e]; av[e + 8] = a1[e];
      bv[e] = b0[e]; bv[e + 8] = b1[e];
    }
    acc = __builtin_amdgcn_wmma_f32_16x16x32_f16(
        false, av, false, bv, (short)0, acc, false, false);
  }

  const int col = n0 + l15;
  const float bb = bias ? bias[col] : 0.0f;
#pragma unroll
  for (int r = 0; r < 8; ++r) {
    int row = m0 + r + hi * 8;                      // C/D layout (ISA 7.12.2)
    float v = acc[r] + bb;
    if (res) v += res[(size_t)row * N + col];
    if (out32) {
      if (trans_out) out32[(size_t)col * M + row] = v;
      else           out32[(size_t)row * N + col] = v;
    }
    if (out16) out16[(size_t)row * N + col] = (h16)v;
  }
}

// ---------------------------------------------------------------------------
// Softmax over NP=4 points, one thread per (query, head).
// aw layout: [n][h][p] contiguous -> float4 per (n,h).
// ---------------------------------------------------------------------------
__global__ void k_softmax4(float* __restrict__ aw) {
  int i = blockIdx.x * 256 + threadIdx.x;          // over NQ*NHD
  if (i >= NQ * NHD) return;
  float4 a = *(const float4*)(aw + (size_t)i * 4);
  float m = fmaxf(fmaxf(a.x, a.y), fmaxf(a.z, a.w));
  float e0 = expf(a.x - m), e1 = expf(a.y - m), e2 = expf(a.z - m), e3 = expf(a.w - m);
  float inv = 1.0f / (e0 + e1 + e2 + e3);
  float4 o = make_float4(e0 * inv, e1 * inv, e2 * inv, e3 * inv);
  *(float4*)(aw + (size_t)i * 4) = o;
}

// ---------------------------------------------------------------------------
// MSDeformAttn bilinear sampling. Thread = (n, h, hd4): 4 HD channels via
// float4 gathers; grid_sample semantics reduce to x_pix = x + off_x (pixel
// centers cancel the -0.5). Zero padding outside [0,128).
// ---------------------------------------------------------------------------
__device__ __forceinline__ void samp_corner(const float* __restrict__ val,
                                            int yi, int xi, int dcol, float w,
                                            float4& acc) {
  if (yi < 0 || yi >= HH || xi < 0 || xi >= WWD) return;
  const float4 g = *(const float4*)(val + ((size_t)(yi * WWD + xi) * DD + dcol));
  acc.x += w * g.x; acc.y += w * g.y; acc.z += w * g.z; acc.w += w * g.w;
}

__global__ void k_sample(const float* __restrict__ val, const float* __restrict__ off,
                         const float* __restrict__ aw, h16* __restrict__ out) {
  int i = blockIdx.x * 256 + threadIdx.x;          // over NQ * NHD * (HDIM/4)
  if (i >= NQ * NHD * (HDIM / 4)) return;
  int n   = i >> 6;
  int t   = i & 63;
  int h   = t >> 3;
  int hd4 = t & 7;
  int y = n >> 7, x = n & 127;
  int dcol = h * HDIM + hd4 * 4;

  const float* offp = off + (size_t)n * (NHD * NPT * 2) + h * (NPT * 2);
  const float* awp  = aw  + (size_t)n * (NHD * NPT)     + h * NPT;

  float4 acc = make_float4(0.f, 0.f, 0.f, 0.f);
#pragma unroll
  for (int p = 0; p < NPT; ++p) {
    float xf = (float)x + offp[p * 2 + 0];
    float yf = (float)y + offp[p * 2 + 1];
    float w  = awp[p];
    float x0f = floorf(xf), y0f = floorf(yf);
    float lx = xf - x0f, ly = yf - y0f;
    int xi = (int)x0f, yi = (int)y0f;
    samp_corner(val, yi,     xi,     dcol, w * (1.f - lx) * (1.f - ly), acc);
    samp_corner(val, yi,     xi + 1, dcol, w * lx * (1.f - ly),         acc);
    samp_corner(val, yi + 1, xi,     dcol, w * (1.f - lx) * ly,         acc);
    samp_corner(val, yi + 1, xi + 1, dcol, w * lx * ly,                 acc);
  }
  v4h o; o[0] = (h16)acc.x; o[1] = (h16)acc.y; o[2] = (h16)acc.z; o[3] = (h16)acc.w;
  *(v4h*)(out + (size_t)n * DD + dcol) = o;
}

// ---------------------------------------------------------------------------
// LayerNorm + ReLU, one wave32 per row (L = 128 or 64), f16 output.
// ---------------------------------------------------------------------------
__global__ __launch_bounds__(256)
void k_lnrelu(const float* __restrict__ in, const float* __restrict__ g,
              const float* __restrict__ b, h16* __restrict__ out, int L) {
  int wave = threadIdx.x >> 5, lane = threadIdx.x & 31;
  int row  = blockIdx.x * 8 + wave;
  if (row >= NQ) return;
  const float* rp = in + (size_t)row * L;
  int E = L >> 5;                                   // elems per lane (<=4)
  float x[4];
  float s = 0.f;
  for (int e = 0; e < E; ++e) { x[e] = rp[lane + e * 32]; s += x[e]; }
  for (int o = 16; o; o >>= 1) s += __shfl_xor(s, o, 32);
  float mu = s / (float)L;
  float v2 = 0.f;
  for (int e = 0; e < E; ++e) { float d = x[e] - mu; v2 += d * d; }
  for (int o = 16; o; o >>= 1) v2 += __shfl_xor(v2, o, 32);
  float inv = rsqrtf(v2 / (float)L + 1e-5f);
  for (int e = 0; e < E; ++e) {
    int c = lane + e * 32;
    float yv = (x[e] - mu) * inv * g[c] + b[c];
    out[(size_t)row * L + c] = (h16)fmaxf(yv, 0.f);
  }
}

// ---------------------------------------------------------------------------
// Host orchestration
// ---------------------------------------------------------------------------
extern "C" void kernel_launch(void* const* d_in, const int* in_sizes, int n_in,
                              void* d_out, int out_size, void* d_ws, size_t ws_size,
                              hipStream_t stream) {
  const float* query    = (const float*)d_in[0];
  const float* value    = (const float*)d_in[1];
  const float* conv_q_w = (const float*)d_in[2];
  const float* conv_q_b = (const float*)d_in[3];
  const float* conv_v_w = (const float*)d_in[4];
  const float* conv_v_b = (const float*)d_in[5];
  const float* row_emb  = (const float*)d_in[6];
  const float* col_emb  = (const float*)d_in[7];
  const float* lin_q_w  = (const float*)d_in[8];
  const float* lin_q_b  = (const float*)d_in[9];
  const float* lin_v_w  = (const float*)d_in[10];
  const float* lin_v_b  = (const float*)d_in[11];
  const float* off_w    = (const float*)d_in[12];
  const float* off_b    = (const float*)d_in[13];
  const float* aw_w     = (const float*)d_in[14];
  const float* aw_b     = (const float*)d_in[15];
  const float* vp_w     = (const float*)d_in[16];
  const float* vp_b     = (const float*)d_in[17];
  const float* op_w     = (const float*)d_in[18];
  const float* op_b     = (const float*)d_in[19];
  const float* out_w1   = (const float*)d_in[20];
  const float* out_b1   = (const float*)d_in[21];
  const float* ln1_g    = (const float*)d_in[22];
  const float* ln1_b    = (const float*)d_in[23];
  const float* out_w2   = (const float*)d_in[24];
  const float* out_b2   = (const float*)d_in[25];
  const float* ln2_g    = (const float*)d_in[26];
  const float* ln2_b    = (const float*)d_in[27];
  const float* out_w3   = (const float*)d_in[28];
  const float* out_b3   = (const float*)d_in[29];
  float* outp = (float*)d_out;

  // ---- workspace layout (overlapped by lifetime) ----
  constexpr size_t SZ_AQ    = (size_t)NQ * CINC * 2;   // 12.6 MB
  constexpr size_t OFF_AQ   = 0;
  constexpr size_t OFF_AV   = OFF_AQ + SZ_AQ;
  constexpr size_t OFF_QP16 = OFF_AQ;                  // reuse after conv GEMMs
  constexpr size_t OFF_QT16 = OFF_AV + SZ_AQ;          // qtmp f16, 8 MB
  constexpr size_t OFF_VT16 = OFF_QT16 + (size_t)NQ * DD * 2;
  constexpr size_t OFF_OFF32 = OFF_QT16;               // reuse, 4 MB
  constexpr size_t OFF_AW32  = OFF_OFF32 + (size_t)NQ * 64 * 4;
  constexpr size_t OFF_SAMP16 = OFF_VT16;              // reuse, 8 MB
  constexpr size_t OFF_Q32  = OFF_VT16 + (size_t)NQ * DD * 2;   // 16 MB
  constexpr size_t OFF_Q16  = OFF_Q32 + (size_t)NQ * DD * 4;    // 8 MB
  constexpr size_t OFF_V16  = OFF_Q16 + (size_t)NQ * DD * 2;    // 8 MB
  constexpr size_t OFF_POS32 = OFF_V16 + (size_t)NQ * DD * 2;   // 16 MB
  constexpr size_t OFF_VAL32 = OFF_POS32 + (size_t)NQ * DD * 4; // 16 MB
  constexpr size_t OFF_H1_32  = OFF_VAL32;             // MLP phase reuse
  constexpr size_t OFF_H1LN16 = OFF_H1_32 + (size_t)NQ * 128 * 4;
  constexpr size_t OFF_H2_32  = OFF_H1LN16 + (size_t)NQ * 128 * 2;
  constexpr size_t OFF_H2LN16 = OFF_V16;               // v16 dead in MLP phase
  constexpr size_t OFF_W    = OFF_VAL32 + (size_t)NQ * DD * 4;
  // f16 weight sub-offsets (in halfs)
  constexpr size_t W_WQ = 0, W_WV = 98304, W_LQ = 196608, W_LV = 262144;
  constexpr size_t W_L0 = 327680, W_LSTR = 155648;     // +0 off, +16384 aw, +24576 vp, +90112 op
  constexpr size_t W_M1 = 950272, W_M2 = 983040, W_M3 = 991232;

  char* ws = (char*)d_ws;
  auto Hp = [&](size_t off) { return (h16*)(ws + off); };
  auto Fp = [&](size_t off) { return (float*)(ws + off); };
  h16* WB = Hp(OFF_W);

  auto cvt = [&](const float* in, h16* out, int R, int C, int trans) {
    int tot = R * C;
    k_cvt<<<(tot + 255) / 256, 256, 0, stream>>>(in, out, R, C, trans);
  };
  auto gemm = [&](const h16* A, const h16* Bt, const float* bias, const float* res,
                  float* o32, h16* o16, int M, int N, int K, int trans) {
    dim3 grid(M / 16, N / 16);
    k_gemm<<<grid, 32, 0, stream>>>(A, Bt, bias, res, o32, o16, M, N, K, trans);
  };

  // ---- prep: f16 activations (transposed) + f16 Bt weights + pos ----
  cvt(query, Hp(OFF_AQ), CINC, NQ, 1);               // (384 x Nq) -> (Nq x 384)
  cvt(value, Hp(OFF_AV), CINC, NQ, 1);
  cvt(conv_q_w, WB + W_WQ, DD, CINC, 0);             // already (out=N) x (in=K)
  cvt(conv_v_w, WB + W_WV, DD, CINC, 0);
  cvt(lin_q_w, WB + W_LQ, DD, DD, 1);                // (K x N) -> (N x K)
  cvt(lin_v_w, WB + W_LV, DD, DD, 1);
  for (int i = 0; i < 4; ++i) {
    size_t base = W_L0 + (size_t)i * W_LSTR;
    cvt(off_w + (size_t)i * DD * 64, WB + base,          DD, 64,  1);
    cvt(aw_w  + (size_t)i * DD * 32, WB + base + 16384,  DD, 32,  1);
    cvt(vp_w  + (size_t)i * DD * DD, WB + base + 24576,  DD, DD,  1);
    cvt(op_w  + (size_t)i * DD * DD, WB + base + 90112,  DD, DD,  1);
  }
  cvt(out_w1, WB + W_M1, DD, 128, 1);
  cvt(out_w2, WB + W_M2, 128, 64, 1);
  cvt(out_w3, WB + W_M3, 64, CINC, 1);
  k_pos<<<(NQ * DD) / 256, 256, 0, stream>>>(row_emb, col_emb, Fp(OFF_POS32));

  // ---- input projections ----
  gemm(Hp(OFF_AQ), WB + W_WQ, conv_q_b, nullptr, nullptr, Hp(OFF_QT16), NQ, DD, CINC, 0);
  gemm(Hp(OFF_AV), WB + W_WV, conv_v_b, nullptr, nullptr, Hp(OFF_VT16), NQ, DD, CINC, 0);
  gemm(Hp(OFF_QT16), WB + W_LQ, lin_q_b, nullptr, Fp(OFF_Q32), Hp(OFF_Q16), NQ, DD, DD, 0);
  gemm(Hp(OFF_VT16), WB + W_LV, lin_v_b, nullptr, nullptr,     Hp(OFF_V16), NQ, DD, DD, 0);

  // ---- 4 MSDeformAttn layers (3 cross + 1 self) ----
  for (int i = 0; i < 4; ++i) {
    size_t base = W_L0 + (size_t)i * W_LSTR;
    const h16* val16 = (i < 3) ? Hp(OFF_V16) : Hp(OFF_Q16);
    k_addpos<<<(NQ * DD) / 256, 256, 0, stream>>>(Fp(OFF_Q32), Fp(OFF_POS32), Hp(OFF_QP16));
    gemm(val16,        WB + base + 24576, vp_b + (size_t)i * DD, nullptr,
         Fp(OFF_VAL32), nullptr, NQ, DD, DD, 0);
    gemm(Hp(OFF_QP16), WB + base,          off_b + (size_t)i * 64, nullptr,
         Fp(OFF_OFF32), nullptr, NQ, 64, DD, 0);
    gemm(Hp(OFF_QP16), WB + base + 16384,  aw_b + (size_t)i * 32, nullptr,
         Fp(OFF_AW32), nullptr, NQ, 32, DD, 0);
    k_softmax4<<<(NQ * NHD) / 256, 256, 0, stream>>>(Fp(OFF_AW32));
    k_sample<<<(NQ * NHD * (HDIM / 4)) / 256, 256, 0, stream>>>(
        Fp(OFF_VAL32), Fp(OFF_OFF32), Fp(OFF_AW32), Hp(OFF_SAMP16));
    gemm(Hp(OFF_SAMP16), WB + base + 90112, op_b + (size_t)i * DD, Fp(OFF_Q32),
         Fp(OFF_Q32), Hp(OFF_Q16), NQ, DD, DD, 0);   // in-place residual
  }

  // ---- output MLP: 256 -> 128 -> 64 -> 384 with LN+ReLU ----
  gemm(Hp(OFF_Q16), WB + W_M1, out_b1, nullptr, Fp(OFF_H1_32), nullptr, NQ, 128, DD, 0);
  k_lnrelu<<<NQ / 8, 256, 0, stream>>>(Fp(OFF_H1_32), ln1_g, ln1_b, Hp(OFF_H1LN16), 128);
  gemm(Hp(OFF_H1LN16), WB + W_M2, out_b2, nullptr, Fp(OFF_H2_32), nullptr, NQ, 64, 128, 0);
  k_lnrelu<<<NQ / 8, 256, 0, stream>>>(Fp(OFF_H2_32), ln2_g, ln2_b, Hp(OFF_H2LN16), 64);
  // final GEMM stores transposed directly in BEV layout (Cin, H, W)
  gemm(Hp(OFF_H2LN16), WB + W_M3, out_b3, nullptr, outp, nullptr, NQ, CINC, 64, 1);
}
